// GATNet_85693187489803
// MI455X (gfx1250) — matrix operand order repaired and checked by reading, not compiled, not run on hardware
//
#include <hip/hip_runtime.h>
#include <hip/hip_bf16.h>
#include <hip/hip_fp16.h>

typedef __attribute__((ext_vector_type(16))) _Float16 v16h;
typedef __attribute__((ext_vector_type(8)))  _Float16 v8h;
typedef __attribute__((ext_vector_type(8)))  float    v8f;

#define HID_   128
#define HEADS_ 4

// ---- order-preserving float<->uint encoding for atomic segment-max ----
__device__ __forceinline__ unsigned f2ord(float f) {
  unsigned u = __float_as_uint(f);
  return (u & 0x80000000u) ? ~u : (u | 0x80000000u);
}
__device__ __forceinline__ float ord2f(unsigned e) {
  return (e & 0x80000000u) ? __uint_as_float(e & 0x7fffffffu) : __uint_as_float(~e);
}

// ---------------------------------------------------------------- utils
__global__ void k_cvt_f16(const float* __restrict__ s, _Float16* __restrict__ d, long n) {
  long i = (long)blockIdx.x * blockDim.x + threadIdx.x;
  if (i < n) d[i] = (_Float16)s[i];
}

__global__ void k_fill_u32(unsigned* __restrict__ p, unsigned v, long n) {
  long i = (long)blockIdx.x * blockDim.x + threadIdx.x;
  if (i < n) p[i] = v;
}

__global__ void k_add_bias(float* __restrict__ o, const float* __restrict__ b, long n, int mask) {
  long i = (long)blockIdx.x * blockDim.x + threadIdx.x;
  if (i < n) o[i] += b[(int)i & mask];
}

// ---------------------------------------------------------------- WMMA GEMM
// Y[M,Nc] = A[M,K](f16) * B[Nc,K](f16, k-contiguous, i.e. W row-major)^T  (+bias)
// blockDim = 512 (16 waves). One wave -> one 16x16 output tile.
__global__ void k_gemm_wmma(const _Float16* __restrict__ A, const _Float16* __restrict__ B,
                            float* __restrict__ Y, const float* __restrict__ bias,
                            int M, int K, int Nc) {
  const int lane = threadIdx.x & 31;
  const int wave = threadIdx.x >> 5;                  // 0..15
  const int nTiles = Nc >> 4;
  const int wavesPerRow = nTiles < 16 ? nTiles : 16;
  const int mPerBlock = 16 / wavesPerRow;
  const int mTile = blockIdx.x * mPerBlock + wave / wavesPerRow;
  if (mTile * 16 >= M) return;                        // wave-uniform; EXEC stays all-1 for WMMA

  const int l15 = lane & 15;
  const int hs  = lane >> 4;                          // half-select (lanes 16-31)
  const int m0  = mTile * 16;

  // A fragment addressing per ISA 16-bit A layout:
  // lane<16: K = {k0..k0+7} in v0-3, {k0+16..k0+23} in v4-7 ; lane>=16 shifted by 8.
  const _Float16* aBase = A + (size_t)(m0 + l15) * K + hs * 8;

  for (int nTile = wave % wavesPerRow; nTile < nTiles; nTile += wavesPerRow) {
    const int n0 = nTile * 16;
    // B fragment: lane<16 -> N=n0+lane, K={k0..k0+7,k0+16..k0+23}; lane>=16 shifted by 8.
    const _Float16* bBase = B + (size_t)(n0 + l15) * K + hs * 8;
    v8f c = {};
    for (int k0 = 0; k0 < K; k0 += 32) {
      union { v16h v; v8h h[2]; } a;
      a.h[0] = *(const v8h*)(aBase + k0);
      a.h[1] = *(const v8h*)(aBase + k0 + 16);
      union { v16h v; v8h h[2]; } b;
      b.h[0] = *(const v8h*)(bBase + k0);
      b.h[1] = *(const v8h*)(bBase + k0 + 16);
      c = __builtin_amdgcn_wmma_f32_16x16x32_f16(false, a.v, false, b.v,
                                                 (short)0, c, false, false);
    }
    // D layout: vgpr r, lane<16 -> (M=r, N=lane); lane>=16 -> (M=8+r, N=lane-16)
    const float bv = bias ? bias[n0 + l15] : 0.0f;
    float* yBase = Y + (size_t)(m0 + hs * 8) * Nc + n0 + l15;
#pragma unroll
    for (int r = 0; r < 8; ++r) yBase[(size_t)r * Nc] = c[r] + bv;
  }
}

// ---------------------------------------------------------------- edge passes
// xlxr: [N,256] f32 ; row = [xl(128) | xr(128)]
__global__ void k_edge_logits(const float* __restrict__ xlxr, const int* __restrict__ ei,
                              int E_, int N_, const float* __restrict__ att,
                              float* __restrict__ lg, unsigned* __restrict__ mOrd) {
  const int e = blockIdx.x * (blockDim.x >> 5) + (threadIdx.x >> 5);
  const int lane = threadIdx.x & 31;
  if (e >= E_ + N_) return;
  const int s = (e < E_) ? ei[e] : (e - E_);
  const int d = (e < E_) ? ei[(size_t)E_ + e] : (e - E_);
  const float* xl = xlxr + (size_t)s * 256;
  const float* xr = xlxr + (size_t)d * 256 + 128;
  float acc[HEADS_];
#pragma unroll
  for (int h = 0; h < HEADS_; ++h) {
    const int c = h * 32 + lane;
    float v = xl[c] + xr[c];
    v = (v > 0.0f) ? v : 0.2f * v;                    // leaky_relu
    acc[h] = v * att[c];
  }
#pragma unroll
  for (int off = 16; off > 0; off >>= 1)
#pragma unroll
    for (int h = 0; h < HEADS_; ++h) acc[h] += __shfl_xor(acc[h], off, 32);
  if (lane == 0) {
#pragma unroll
    for (int h = 0; h < HEADS_; ++h) {
      lg[(size_t)e * HEADS_ + h] = acc[h];
      atomicMax(mOrd + (size_t)d * HEADS_ + h, f2ord(acc[h]));
    }
  }
}

__global__ void k_edge_expsum(float* __restrict__ lg, const int* __restrict__ ei,
                              int E_, int N_, const unsigned* __restrict__ mOrd,
                              float* __restrict__ den) {
  const long t = (long)blockIdx.x * blockDim.x + threadIdx.x;
  if (t >= (long)(E_ + N_) * HEADS_) return;
  const int e = (int)(t >> 2);
  const int h = (int)(t & 3);
  const int d = (e < E_) ? ei[(size_t)E_ + e] : (e - E_);
  const float ex = __expf(lg[t] - ord2f(mOrd[(size_t)d * HEADS_ + h]));
  lg[t] = ex;                                         // overwrite logits with exp values
  atomicAdd(den + (size_t)d * HEADS_ + h, ex);
}

__global__ void k_edge_aggregate(const float* __restrict__ ex, const float* __restrict__ den,
                                 const float* __restrict__ xlxr, const int* __restrict__ ei,
                                 int E_, int N_, float* __restrict__ out) {
  const int e = blockIdx.x * (blockDim.x >> 5) + (threadIdx.x >> 5);
  const int lane = threadIdx.x & 31;
  if (e >= E_ + N_) return;
  const int s = (e < E_) ? ei[e] : (e - E_);
  const int d = (e < E_) ? ei[(size_t)E_ + e] : (e - E_);
  const float* xl = xlxr + (size_t)s * 256;
  float* o = out + (size_t)d * HID_;
#pragma unroll
  for (int h = 0; h < HEADS_; ++h) {
    const float alpha = ex[(size_t)e * HEADS_ + h] / den[(size_t)d * HEADS_ + h];
    const int c = h * 32 + lane;
    atomicAdd(o + c, alpha * xl[c]);
  }
}

// ---------------------------------------------------------------- driver
static inline int cdiv_l(long a, long b) { return (int)((a + b - 1) / b); }

extern "C" void kernel_launch(void* const* d_in, const int* in_sizes, int n_in,
                              void* d_out, int out_size, void* d_ws, size_t ws_size,
                              hipStream_t stream) {
  (void)n_in; (void)out_size; (void)ws_size;
  const float* x    = (const float*)d_in[0];
  const int*   ei   = (const int*)d_in[1];
  const float* Wl0  = (const float*)d_in[2];
  const float* Wr0  = (const float*)d_in[3];
  const float* att0 = (const float*)d_in[4];
  const float* b0   = (const float*)d_in[5];
  const float* Wl1  = (const float*)d_in[6];
  const float* Wr1  = (const float*)d_in[7];
  const float* att1 = (const float*)d_in[8];
  const float* b1   = (const float*)d_in[9];
  const float* dW   = (const float*)d_in[10];
  const float* db   = (const float*)d_in[11];
  float* out = (float*)d_out;

  const int DIN = 256;
  const int N = in_sizes[0] / DIN;        // 50000 (multiple of 16)
  const int E = in_sizes[1] / 2;          // 800000
  const int Etot = E + N;

  // ---- workspace layout (all offsets 32B-aligned) ----
  char* ws = (char*)d_ws;
  size_t oXLXR = 0;                                   // [N,256] f32
  size_t oF16  = oXLXR + (size_t)N * 256 * 4;         // [N,256] f16 (reused for h f16)
  size_t oH    = oF16  + (size_t)N * 256 * 2;         // [N,128] f32
  size_t oLg   = oH    + (size_t)N * HID_ * 4;        // [Etot,4] f32
  size_t oM    = oLg   + (size_t)Etot * HEADS_ * 4;   // [N,4] u32 (ordered max)
  size_t oDen  = oM    + (size_t)N * HEADS_ * 4;      // [N,4] f32
  size_t oW0   = oDen  + (size_t)N * HEADS_ * 4;      // [256,256] f16
  size_t oW1   = oW0   + (size_t)256 * 256 * 2;       // [256,128] f16
  size_t oWd   = oW1   + (size_t)256 * 128 * 2;       // [32,128] f16

  float*     xlxr = (float*)(ws + oXLXR);
  _Float16*  af16 = (_Float16*)(ws + oF16);
  float*     hbuf = (float*)(ws + oH);
  float*     lg   = (float*)(ws + oLg);
  unsigned*  mOrd = (unsigned*)(ws + oM);
  float*     den  = (float*)(ws + oDen);
  _Float16*  w0   = (_Float16*)(ws + oW0);
  _Float16*  w1   = (_Float16*)(ws + oW1);
  _Float16*  wd   = (_Float16*)(ws + oWd);

  const int mTiles = N / 16;
  const long nh = (long)N * HEADS_;
  const long nhid = (long)N * HID_;
  const unsigned NEG_INF_ORD = 0x007FFFFFu;           // f2ord(-inf)

  // ---- weight conversions (stack [Wl;Wr] along output dim) ----
  k_cvt_f16<<<cdiv_l(128L*256, 256), 256, 0, stream>>>(Wl0, w0,            128L*256);
  k_cvt_f16<<<cdiv_l(128L*256, 256), 256, 0, stream>>>(Wr0, w0 + 128*256,  128L*256);
  k_cvt_f16<<<cdiv_l(128L*128, 256), 256, 0, stream>>>(Wl1, w1,            128L*128);
  k_cvt_f16<<<cdiv_l(128L*128, 256), 256, 0, stream>>>(Wr1, w1 + 128*128,  128L*128);
  k_cvt_f16<<<cdiv_l(32L*128,  256), 256, 0, stream>>>(dW,  wd,            32L*128);

  // ================= layer 0 =================
  k_cvt_f16<<<cdiv_l((long)N * 256, 256), 256, 0, stream>>>(x, af16, (long)N * 256);
  k_gemm_wmma<<<mTiles, 512, 0, stream>>>(af16, w0, xlxr, nullptr, N, 256, 256);
  k_fill_u32<<<cdiv_l(nh, 256), 256, 0, stream>>>(mOrd, NEG_INF_ORD, nh);
  k_fill_u32<<<cdiv_l(nh, 256), 256, 0, stream>>>((unsigned*)den, 0u, nh);
  k_edge_logits<<<cdiv_l(Etot, 8), 256, 0, stream>>>(xlxr, ei, E, N, att0, lg, mOrd);
  k_edge_expsum<<<cdiv_l((long)Etot * HEADS_, 256), 256, 0, stream>>>(lg, ei, E, N, mOrd, den);
  k_fill_u32<<<cdiv_l(nhid, 256), 256, 0, stream>>>((unsigned*)hbuf, 0u, nhid);
  k_edge_aggregate<<<cdiv_l(Etot, 8), 256, 0, stream>>>(lg, den, xlxr, ei, E, N, hbuf);
  k_add_bias<<<cdiv_l(nhid, 256), 256, 0, stream>>>(hbuf, b0, nhid, HID_ - 1);

  // ================= layer 1 =================
  k_cvt_f16<<<cdiv_l(nhid, 256), 256, 0, stream>>>(hbuf, af16, nhid);
  k_gemm_wmma<<<mTiles, 512, 0, stream>>>(af16, w1, xlxr, nullptr, N, 128, 256);
  k_fill_u32<<<cdiv_l(nh, 256), 256, 0, stream>>>(mOrd, NEG_INF_ORD, nh);
  k_fill_u32<<<cdiv_l(nh, 256), 256, 0, stream>>>((unsigned*)den, 0u, nh);
  k_edge_logits<<<cdiv_l(Etot, 8), 256, 0, stream>>>(xlxr, ei, E, N, att1, lg, mOrd);
  k_edge_expsum<<<cdiv_l((long)Etot * HEADS_, 256), 256, 0, stream>>>(lg, ei, E, N, mOrd, den);
  k_fill_u32<<<cdiv_l(nhid, 256), 256, 0, stream>>>((unsigned*)hbuf, 0u, nhid);
  k_edge_aggregate<<<cdiv_l(Etot, 8), 256, 0, stream>>>(lg, den, xlxr, ei, E, N, hbuf);
  k_add_bias<<<cdiv_l(nhid, 256), 256, 0, stream>>>(hbuf, b1, nhid, HID_ - 1);

  // ================= down proj =================
  k_cvt_f16<<<cdiv_l(nhid, 256), 256, 0, stream>>>(hbuf, af16, nhid);
  // Nc=32 -> 2 n-tiles per wave pair, 8 m-tiles per block
  k_gemm_wmma<<<cdiv_l(mTiles, 8), 512, 0, stream>>>(af16, wd, out, db, N, 128, 32);
}
